// AttentionBlock_53455162966276
// MI455X (gfx1250) — matrix operand order repaired and checked by reading, not compiled, not run on hardware
//
#include <hip/hip_runtime.h>

typedef __attribute__((ext_vector_type(16))) __bf16 v16bf;
typedef __attribute__((ext_vector_type(8)))  float  v8f;

#define NB 2
#define NT 2048
#define ND 256
#define NH 8

__device__ __forceinline__ v8f wmma_bf16(v16bf a, v16bf b, v8f c) {
    // D = A(16x32 bf16) * B(32x16 bf16) + C(16x16 f32)
    return __builtin_amdgcn_wmma_f32_16x16x32_bf16(
        /*neg_a=*/false, a, /*neg_b=*/false, b,
        /*c_mod=*/(short)0, c, /*reuse_a=*/false, /*reuse_b=*/false);
}

// ---------------------------------------------------------------------------
// f32 -> bf16 conversion: Kbf = bf16(r'), Qbf = bf16(Q), ETbf[h][j][i] = bf16(E[h][i][j])
// ---------------------------------------------------------------------------
__global__ void convert_kernel(const float* __restrict__ r,
                               const float* __restrict__ Q,
                               const float* __restrict__ E,
                               __bf16* __restrict__ Kbf,
                               __bf16* __restrict__ Qbf,
                               __bf16* __restrict__ ETbf) {
    int idx = blockIdx.x * 256 + threadIdx.x;
    if (idx < NB * NT * ND) Kbf[idx] = (__bf16)r[idx];
    if (idx < NH * ND * ND) {
        Qbf[idx] = (__bf16)Q[idx];
        int h = idx >> 16;
        int rem = idx & 0xFFFF;
        int i = rem >> 8;
        int j = rem & 0xFF;
        ETbf[(h << 16) + (j << 8) + i] = (__bf16)E[idx];
    }
}

// ---------------------------------------------------------------------------
// C[b,h] (NT x ND) = X[b] (NT x ND) @ W[h] (ND x ND), all bf16.
// TR=false: store row-major Out[bh][m][n]   (used for A = r'.Q)
// TR=true : store transposed Out[bh][n][m]  (used for Vt[i][u])
// grid = (NT/16, ND/16, NB*NH), block = 32 (one wave per 16x16 output tile)
// ---------------------------------------------------------------------------
template <bool TR>
__global__ __launch_bounds__(32) void gemm16_kernel(const __bf16* __restrict__ X,
                                                    const __bf16* __restrict__ W,
                                                    __bf16* __restrict__ Out) {
    int mt = blockIdx.x, nt = blockIdx.y, bh = blockIdx.z;
    int b = bh >> 3, h = bh & 7;
    int lane = threadIdx.x & 31;
    int lm = lane & 15;           // row/col within 16
    int lhi = lane >> 4;          // lane group

    const __bf16* Xb = X + (size_t)b * NT * ND + (size_t)(mt * 16) * ND;
    const __bf16* Wh = W + (size_t)h * ND * ND;

    v8f acc = {};
#pragma unroll
    for (int kc = 0; kc < 8; ++kc) {
        v16bf a, bm;
#pragma unroll
        for (int e = 0; e < 16; ++e) {
            // A 16x32 as loaded here (contiguous k, same as B): M = lane&15, K = e (+16 for upper lanes)
            int ka = e + (lhi ? 16 : 0);
            a[e] = Xb[(size_t)lm * ND + kc * 32 + ka];
            // B 32x16 as loaded here (contiguous k, same as A): N = lane&15, K = e (+16 for upper lanes)
            int kb = kc * 32 + e + (lhi ? 16 : 0);
            bm[e] = Wh[(size_t)kb * ND + nt * 16 + lm];
        }
        acc = wmma_bf16(a, bm, acc);
    }

    if (TR) {
        // Out[bh][n][m]: per lane the 8 C values are contiguous along m -> b128 store
        __bf16* ob = Out + (size_t)bh * ND * NT;
#pragma unroll
        for (int c = 0; c < 8; ++c)
            ob[(size_t)(nt * 16 + lm) * NT + mt * 16 + 8 * lhi + c] = (__bf16)acc[c];
    } else {
        __bf16* ob = Out + (size_t)bh * NT * ND;
#pragma unroll
        for (int c = 0; c < 8; ++c) {
            int m = c + 8 * lhi; // C layout: M = vgpr + 8*(lane>=16), N = lane&15
            ob[(size_t)(mt * 16 + m) * ND + nt * 16 + lm] = (__bf16)acc[c];
        }
    }
}

// ---------------------------------------------------------------------------
// Flash-style fused kernel:
//   for each (b, t-tile of 16): S = A_h . K^T (causal), O += S . V_h, sum over h.
// block = 256 threads = 8 waves; wave w handles head h = w.
// K u-chunks are double-buffered in LDS via GLOBAL_LOAD_ASYNC_TO_LDS (ASYNCcnt):
// chunk i+1's DMA is issued before computing chunk i, hiding the fill latency
// under 32 WMMAs. One barrier per iteration. grid.x = NB * (NT/16)
// ---------------------------------------------------------------------------
#define KPAD 8   // row padding (bf16 elems): row stride 528B == 4 dwords mod 64 banks
#define SPAD 8
#define KROW (ND + KPAD)

__global__ __launch_bounds__(256) void flash_kernel(const __bf16* __restrict__ A,
                                                    const __bf16* __restrict__ Kbf,
                                                    const __bf16* __restrict__ Vt,
                                                    float* __restrict__ out) {
    __shared__ float  red[16 * 256];            // head-reduction buffer (16 KB)
    __shared__ __bf16 kTile[2][32][KROW];       // double-buffered K chunk (33 KB)
    __shared__ __bf16 sbuf[8][16][32 + SPAD];   // per-wave S bounce (10 KB)

    int tid = threadIdx.x;
    int lane = tid & 31;
    int w = tid >> 5;                           // wave id == head id
    int b = blockIdx.x / (NT / 16);
    int t0 = (blockIdx.x % (NT / 16)) * 16;

    for (int i = tid; i < 16 * 256; i += 256) red[i] = 0.0f;

    int lm = lane & 15;
    int lhi = lane >> 4;

    // Preload query-side fragments A[b,h, t0..t0+15, 0..255] (8 chunks of K=32)
    const __bf16* Abase = A + ((size_t)(b * NH + w) * NT + t0) * ND;
    v16bf afrag[8];
#pragma unroll
    for (int kc = 0; kc < 8; ++kc) {
#pragma unroll
        for (int e = 0; e < 16; ++e) {
            int k = e + (lhi ? 16 : 0);
            afrag[kc][e] = Abase[(size_t)lm * ND + kc * 32 + k];
        }
    }

    v8f acc[16];
    {
        v8f z = {};
#pragma unroll
        for (int n = 0; n < 16; ++n) acc[n] = z;
    }

    const __bf16* Kb  = Kbf + (size_t)b * NT * ND;
    const __bf16* Vtb = Vt + (size_t)(b * NH + w) * ND * NT;

    // Cooperative K-tile DMA mapping: 8 threads per row, 64 contiguous bytes each
    // (a 64B segment never crosses the padded LDS row boundary).
    int krow = tid >> 3;
    int kcol = (tid & 7) * 32;
    const unsigned KTILE_BYTES = 32u * KROW * 2u;
    unsigned ldsSeg = (unsigned)(size_t)(&kTile[0][0][0])
                    + (unsigned)(krow * KROW + kcol) * 2u;

    // Issue the per-lane cache->LDS DMA for one 32x256 K chunk (ASYNCcnt-tracked).
    // The immediate offset adds to both the LDS and the global address.
    auto issue_k_dma = [&](int buf, int uc) {
        unsigned dst = ldsSeg + (unsigned)buf * KTILE_BYTES;
        const __bf16* src = Kb + (size_t)(uc + krow) * ND + kcol;
        asm volatile(
            "global_load_async_to_lds_b128 %0, %1, off\n\t"
            "global_load_async_to_lds_b128 %0, %1, off offset:16\n\t"
            "global_load_async_to_lds_b128 %0, %1, off offset:32\n\t"
            "global_load_async_to_lds_b128 %0, %1, off offset:48"
            :: "v"(dst), "v"(src) : "memory");
    };

    int nChunks = t0 / 32 + 1;
    issue_k_dma(0, 0);                          // prologue DMA for chunk 0

    for (int ci = 0; ci < nChunks; ++ci) {
        int uc = ci * 32;
        int cur = ci & 1;

        // Wait for this wave's DMA of chunk ci, then publish LDS to all waves.
        // The barrier also guarantees every wave's reads of buffer cur^1 (from
        // iteration ci-1) are complete, so it is safe to refill it next.
        asm volatile("s_wait_asynccnt 0" ::: "memory");
        __syncthreads();

        if (ci + 1 < nChunks) issue_k_dma(cur ^ 1, uc + 32);  // overlap next fill

        const __bf16(*kt)[KROW] = kTile[cur];

        // ---- S[t, uc..uc+31] = A . K^T : two 16x16 C tiles, K-loop over 256 ----
        v8f s[2];
#pragma unroll
        for (int half = 0; half < 2; ++half) {
            v8f sc = {};
#pragma unroll
            for (int kc = 0; kc < 8; ++kc) {
                v16bf bfrag;
#pragma unroll
                for (int e = 0; e < 16; ++e) {
                    int k = kc * 32 + e + (lhi ? 16 : 0);
                    bfrag[e] = kt[half * 16 + lm][k];
                }
                sc = wmma_bf16(afrag[kc], bfrag, sc);
            }
            s[half] = sc;
        }

        // ---- causal mask + write S (bf16) to LDS in C layout ----
        bool need_mask = (uc + 32 > t0);
#pragma unroll
        for (int half = 0; half < 2; ++half) {
#pragma unroll
            for (int c = 0; c < 8; ++c) {
                int m = c + 8 * lhi;                 // row within t-tile
                int ucol = uc + half * 16 + lm;      // absolute u
                float v = s[half][c];
                if (need_mask && (ucol > t0 + m)) v = 0.0f;
                sbuf[w][m][half * 16 + lm] = (__bf16)v;
            }
        }
        asm volatile("s_wait_dscnt 0" ::: "memory");

        // ---- read S back as 16x32 bf16 A-fragment (two contiguous b128 reads) ----
        v16bf sfrag;
#pragma unroll
        for (int e = 0; e < 16; ++e) {
            int k = e + (lhi ? 16 : 0);
            sfrag[e] = sbuf[w][lm][k];
        }

        // ---- O[16 x 256] += S(16x32) . V(32x256): 16 N-tiles, contiguous Vt loads ----
        bool has_next = (ci + 1 < nChunks);
#pragma unroll
        for (int n = 0; n < 16; ++n) {
            v16bf vfrag;
            const __bf16* vrow = Vtb + (size_t)(n * 16 + lm) * NT + uc + 16 * lhi;
#pragma unroll
            for (int e = 0; e < 16; ++e)
                vfrag[e] = vrow[e];      // B[k=e+16*lhi][n=lm]: contiguous along u
            if (has_next) __builtin_prefetch(vrow + 32, 0, 1);  // next u-chunk
            acc[n] = wmma_bf16(sfrag, vfrag, acc[n]);
        }
    }

    // ---- reduce over heads through LDS, then one coalesced store ----
    __syncthreads();
#pragma unroll
    for (int n = 0; n < 16; ++n) {
#pragma unroll
        for (int c = 0; c < 8; ++c) {
            int m = c + 8 * lhi;
            atomicAdd(&red[m * 256 + n * 16 + lm], acc[n][c]);
        }
    }
    __syncthreads();

    float* ob = out + ((size_t)b * NT + t0) * ND;
    for (int i = tid; i < 16 * 256; i += 256) ob[i] = red[i];
}

// ---------------------------------------------------------------------------
extern "C" void kernel_launch(void* const* d_in, const int* in_sizes, int n_in,
                              void* d_out, int out_size, void* d_ws, size_t ws_size,
                              hipStream_t stream) {
    const float* r_prime = (const float*)d_in[0];
    const float* Q       = (const float*)d_in[1];
    const float* E       = (const float*)d_in[2];
    float* out = (float*)d_out;

    char* ws = (char*)d_ws;
    const size_t KBF_BYTES = (size_t)NB * NT * ND * 2;       // 2 MB
    const size_t QBF_BYTES = (size_t)NH * ND * ND * 2;       // 1 MB
    const size_t AV_BYTES  = (size_t)NB * NH * NT * ND * 2;  // 16 MB each

    __bf16* Kbf   = (__bf16*)(ws);
    __bf16* Qbf   = (__bf16*)(ws + KBF_BYTES);
    __bf16* ETbf  = (__bf16*)(ws + KBF_BYTES + QBF_BYTES);
    __bf16* Abuf  = (__bf16*)(ws + KBF_BYTES + 2 * QBF_BYTES);
    __bf16* Vtbuf = (__bf16*)(ws + KBF_BYTES + 2 * QBF_BYTES + AV_BYTES);

    // 1) convert inputs to bf16 (and transpose E)
    convert_kernel<<<(NB * NT * ND) / 256, 256, 0, stream>>>(r_prime, Q, E, Kbf, Qbf, ETbf);

    // 2) A[b,h] = r'[b] @ Q[h] (row-major) ; Vt[b,h] = (r'[b] @ E[h]^T)^T  (bf16 WMMA)
    dim3 ggrid(NT / 16, ND / 16, NB * NH);
    gemm16_kernel<false><<<ggrid, 32, 0, stream>>>(Kbf, Qbf, Abuf);
    gemm16_kernel<true><<<ggrid, 32, 0, stream>>>(Kbf, ETbf, Vtbuf);

    // 3) fused causal attention + head reduction
    flash_kernel<<<NB * (NT / 16), 256, 0, stream>>>(Abuf, Kbf, Vtbuf, out);
}
